// SpatialBlock_2525440770970
// MI455X (gfx1250) — compile-verified, matrix operand order checked by reading, not run on hardware
//
#include <hip/hip_runtime.h>
#include <hip/hip_bf16.h>
#include <math.h>

typedef __attribute__((ext_vector_type(16))) _Float16 v16h;
typedef __attribute__((ext_vector_type(8)))  float    v8f;

#define HEADS 4
#define WINP  8
#define WIN2  64
#define NF    33
#define CH    48

// ---------------------------------------------------------------------------
// Bias-free LayerNorm over channel dim of NCHW tensor.
// ---------------------------------------------------------------------------
__global__ __launch_bounds__(256)
void ln_biasfree(const float* __restrict__ x, const float* __restrict__ w,
                 float* __restrict__ y, int C, int HW)
{
    long i   = (long)blockIdx.x * blockDim.x + threadIdx.x;
    int  pix = (int)(i % HW);
    int  b   = (int)(i / HW);
    const float* xp = x + (long)b * C * HW + pix;
    float*       yp = y + (long)b * C * HW + pix;
    float s = 0.f, s2 = 0.f;
    for (int c = 0; c < C; ++c) {
        float v = xp[(long)c * HW];
        s += v; s2 += v * v;
    }
    float mean = s / (float)C;
    float var  = s2 / (float)C - mean * mean;
    float inv  = rsqrtf(var + 1e-5f);
    for (int c = 0; c < C; ++c)
        yp[(long)c * HW] = xp[(long)c * HW] * inv * w[c];
}

// ---------------------------------------------------------------------------
// Weight padding kernels (run once per call; tiny).
// ffn_in_w: 510x96 -> 512x96 (zero rows 510,511).  Same row stride => 1:1 copy.
// ---------------------------------------------------------------------------
__global__ __launch_bounds__(256)
void pad_ffn_in_w(const float* __restrict__ w, float* __restrict__ o)
{
    int i = blockIdx.x * 256 + threadIdx.x;
    if (i >= 512 * 96) return;
    o[i] = (i < 510 * 96) ? w[i] : 0.f;
}

// ffn_out_w: 96x255 -> 96x256 (zero column 255).
__global__ __launch_bounds__(256)
void pad_ffn_out_w(const float* __restrict__ w, float* __restrict__ o)
{
    int i = blockIdx.x * 256 + threadIdx.x;
    if (i >= 96 * 256) return;
    int oc = i >> 8, k = i & 255;
    o[i] = (k < 255) ? w[oc * 255 + k] : 0.f;
}

// ---------------------------------------------------------------------------
// 1x1 conv as WMMA GEMM:  Y[oc][n] = sum_c W[oc][c] * X[c][n]  (+ residual)
// Compile-time C (mult of 32) and OC (mult of 32): ZERO guards, K fully
// unrolled, per-lane base pointers with constant offsets.
// Block: 256 threads = 8 waves; block tile OC=32 x N=256; wave: 16x64 strip.
// ---------------------------------------------------------------------------
template<int C, int OC, bool HASR>
__global__ __launch_bounds__(256)
void gemm1x1_wmma(const float* __restrict__ X, const float* __restrict__ Wt,
                  const float* __restrict__ R, float* __restrict__ Y)
{
    constexpr int HW = 65536;
    const int  b  = blockIdx.z;
    const float* Xb = X + (size_t)b * C  * HW;
    float*       Yb = Y + (size_t)b * OC * HW;
    const float* Rb = R + (size_t)b * OC * HW;

    const int tid   = threadIdx.x;
    const int lane  = tid & 31;
    const int wave  = tid >> 5;
    const int ln    = lane & 15;
    const int khalf = (lane >> 4) << 3;   // 0 or 8

    const int nBase  = blockIdx.x * 256 + (wave >> 1) * 64;
    const int ocBase = blockIdx.y * 32  + (wave & 1) * 16;
    const int m      = ocBase + ln;

    // per-lane base pointers; all k offsets below are compile-time constants
    const float* wrow = Wt + (size_t)m * C + khalf;
    const float* xcol = Xb + (size_t)khalf * HW + nBase + ln;

    v8f acc[4] = {v8f{}, v8f{}, v8f{}, v8f{}};

    #pragma unroll
    for (int kb = 0; kb < C; kb += 32) {
        // ---- A fragment: weights row m, K = kb..kb+31 ----
        v16h a;
        #pragma unroll
        for (int j = 0; j < 4; ++j) {
            a[2 * j]         = (_Float16)wrow[kb + 2 * j];
            a[2 * j + 1]     = (_Float16)wrow[kb + 2 * j + 1];
            a[8 + 2 * j]     = (_Float16)wrow[kb + 2 * j + 16];
            a[8 + 2 * j + 1] = (_Float16)wrow[kb + 2 * j + 17];
        }
        // ---- 4 N sub-tiles share the A fragment ----
        #pragma unroll
        for (int i = 0; i < 4; ++i) {
            const float* xc = xcol + i * 16;
            v16h bf;
            #pragma unroll
            for (int j = 0; j < 4; ++j) {
                bf[2 * j]         = (_Float16)xc[(size_t)(kb + 2 * j)      * HW];
                bf[2 * j + 1]     = (_Float16)xc[(size_t)(kb + 2 * j + 1)  * HW];
                bf[8 + 2 * j]     = (_Float16)xc[(size_t)(kb + 2 * j + 16) * HW];
                bf[8 + 2 * j + 1] = (_Float16)xc[(size_t)(kb + 2 * j + 17) * HW];
            }
            acc[i] = __builtin_amdgcn_wmma_f32_16x16x32_f16(
                         false, a, false, bf, (short)0, acc[i], false, false);
        }
    }

    // ---- store (VGPR r: lanes 0-15 -> M=r, lanes 16-31 -> M=r+8) ----
    #pragma unroll
    for (int i = 0; i < 4; ++i) {
        int n = nBase + i * 16 + ln;
        #pragma unroll
        for (int r = 0; r < 8; ++r) {
            int mm = ocBase + r + khalf;
            size_t off = (size_t)mm * HW + n;
            float v = acc[i][r];
            if (HASR) v += Rb[off];
            Yb[off] = v;
        }
    }
}

// ---------------------------------------------------------------------------
// 3x3 depthwise conv, zero pad 1.  CS = channel stride of the buffers
// (may exceed logical channel count C for padded layouts).
// ---------------------------------------------------------------------------
__global__ __launch_bounds__(256)
void dwconv3(const float* __restrict__ x, const float* __restrict__ w,
             float* __restrict__ y, int C, int CS, int H, int W, long total)
{
    long i = (long)blockIdx.x * blockDim.x + threadIdx.x;
    if (i >= total) return;
    int xx = (int)(i % W); long r = i / W;
    int yy = (int)(r % H); r /= H;
    int c  = (int)(r % C);
    int b  = (int)(r / C);
    const float* wp = w + (long)c * 9;
    const float* xp = x + ((long)b * CS + c) * H * W;
    float*       yp = y + ((long)b * CS + c) * H * W;
    float s = 0.f;
    #pragma unroll
    for (int dy = -1; dy <= 1; ++dy)
        #pragma unroll
        for (int dx = -1; dx <= 1; ++dx) {
            int yz = yy + dy, xz = xx + dx;
            if (yz >= 0 && yz < H && xz >= 0 && xz < W)
                s += xp[(long)yz * W + xz] * wp[(dy + 1) * 3 + (dx + 1)];
        }
    yp[(long)yy * W + xx] = s;
}

// ---------------------------------------------------------------------------
// FFT attention over one 8x8 window: one block per (b, head, patch).
// rDFT via 64-entry twiddle table, 33x33 complex attn, row-L2 normalize,
// out = attn . vf^T, irDFT, scatter to (B, 192, H, W).  LDS ~59.6 KB.
// ---------------------------------------------------------------------------
__global__ __launch_bounds__(256)
void attn_fft_kernel(const float* __restrict__ hid, const float* __restrict__ temp,
                     float* __restrict__ out, int H, int W)
{
    __shared__ float twc[64], tws[64];
    __shared__ float spat[CH * WIN2];
    __shared__ float fre[3 * CH * NF];
    __shared__ float fim[3 * CH * NF];
    __shared__ float are[NF * NF], aim[NF * NF];
    __shared__ float rnorm[NF];

    const int tid = threadIdx.x;
    const int NPx = W / WINP;
    const int NP  = NPx * (H / WINP);
    int blk = blockIdx.x;
    int p   = blk % NP;
    int hd  = (blk / NP) % HEADS;
    int b   = blk / (NP * HEADS);
    int ph  = p / NPx, pw = p % NPx;
    int y0  = ph * WINP, x0 = pw * WINP;
    long base = (long)b * 576 * H * W;

    if (tid < 64) {
        float ang = 6.28318530717958647692f * (float)tid / 64.f;
        twc[tid] = cosf(ang);
        tws[tid] = sinf(ang);
    }

    for (int t = 0; t < 3; ++t) {
        __syncthreads();
        for (int i = tid; i < CH * WIN2; i += 256) {
            int c = i >> 6, s = i & 63;
            int py = s >> 3, px = s & 7;
            int ch = t * 192 + hd * CH + c;
            spat[i] = hid[base + (long)ch * H * W + (long)(y0 + py) * W + (x0 + px)];
        }
        __syncthreads();
        for (int i = tid; i < CH * NF; i += 256) {
            int c = i / NF, f = i % NF;
            const float* sp = &spat[c * WIN2];
            float re = 0.f, im = 0.f;
            for (int xk = 0; xk < 64; ++xk) {
                int idx = (f * xk) & 63;
                re += sp[xk] * twc[idx];
                im -= sp[xk] * tws[idx];
            }
            fre[t * CH * NF + i] = re;
            fim[t * CH * NF + i] = im;
        }
    }
    __syncthreads();

    float tscale = temp[hd];
    for (int i = tid; i < NF * NF; i += 256) {
        int f = i / NF, g = i % NF;
        float re = 0.f, im = 0.f;
        for (int c = 0; c < CH; ++c) {
            float qr = fre[c * NF + f],           qi = fim[c * NF + f];
            float kr = fre[CH * NF + c * NF + g], ki = fim[CH * NF + c * NF + g];
            re += qr * kr - qi * ki;
            im += qr * ki + qi * kr;
        }
        are[i] = re * tscale;
        aim[i] = im * tscale;
    }
    __syncthreads();

    if (tid < NF) {
        float s = 0.f;
        for (int g = 0; g < NF; ++g) {
            float r = are[tid * NF + g], ii = aim[tid * NF + g];
            s += r * r + ii * ii;
        }
        rnorm[tid] = rsqrtf(s);
    }
    __syncthreads();
    for (int i = tid; i < NF * NF; i += 256) {
        float rn = rnorm[i / NF];
        are[i] *= rn;
        aim[i] *= rn;
    }
    __syncthreads();

    for (int i = tid; i < CH * NF; i += 256) {
        int c = i / NF, f = i % NF;
        float re = 0.f, im = 0.f;
        for (int g = 0; g < NF; ++g) {
            float ar = are[f * NF + g], ai = aim[f * NF + g];
            float vr = fre[2 * CH * NF + c * NF + g];
            float vi = fim[2 * CH * NF + c * NF + g];
            re += ar * vr - ai * vi;
            im += ar * vi + ai * vr;
        }
        fre[i] = re;
        fim[i] = im;
    }
    __syncthreads();

    long obase = ((long)b * (HEADS * CH) + hd * CH) * H * W;
    for (int i = tid; i < CH * WIN2; i += 256) {
        int c = i >> 6, n = i & 63;
        float s = 0.f;
        for (int f = 0; f <= 32; ++f) {
            float coef = (f == 0 || f == 32) ? 1.f : 2.f;
            int idx = (f * n) & 63;
            s += coef * (fre[c * NF + f] * twc[idx] - fim[c * NF + f] * tws[idx]);
        }
        s *= (1.0f / 64.0f);
        int py = n >> 3, px = n & 7;
        out[obase + (long)c * H * W + (long)(y0 + py) * W + (x0 + px)] = s;
    }
}

// ---------------------------------------------------------------------------
// GELU(x1) * x2 gating.  Input layout (B, 512, HW): x1 = ch[0..254],
// x2 = ch[255..509].  Output layout (B, 256, HW), channel 255 zeroed
// so the following GEMM can use K = 256 unguarded.
// ---------------------------------------------------------------------------
__global__ __launch_bounds__(256)
void gelu_gate(const float* __restrict__ t, float* __restrict__ g, long total)
{
    long i = (long)blockIdx.x * blockDim.x + threadIdx.x;
    if (i >= total) return;
    int pix = (int)(i % 65536); long r = i / 65536;
    int c = (int)(r & 255);
    int b = (int)(r >> 8);
    if (c == 255) { g[i] = 0.f; return; }
    long ib = (long)b * 512 * 65536;
    float a = t[ib + (long)c * 65536 + pix];
    float v = t[ib + (long)(255 + c) * 65536 + pix];
    float ge = 0.5f * a * (1.f + erff(a * 0.70710678118654752440f));
    g[i] = ge * v;
}

// ---------------------------------------------------------------------------
extern "C" void kernel_launch(void* const* d_in, const int* in_sizes, int n_in,
                              void* d_out, int out_size, void* d_ws, size_t ws_size,
                              hipStream_t stream)
{
    const float* x              = (const float*)d_in[0];
    const float* to_hidden_w    = (const float*)d_in[1];
    const float* to_hidden_dw_w = (const float*)d_in[2];
    const float* project_out_w  = (const float*)d_in[3];
    const float* temperature    = (const float*)d_in[4];
    const float* norm1_w        = (const float*)d_in[5];
    const float* norm2_w        = (const float*)d_in[6];
    const float* ffn_in_w       = (const float*)d_in[7];
    const float* ffn_dw_w       = (const float*)d_in[8];
    const float* ffn_out_w      = (const float*)d_in[9];
    float* out = (float*)d_out;

    const int Bn = 2, DIM = 96, H = 256, W = 256, HW = H * W;

    float* ws      = (float*)d_ws;
    float* bufLN   = ws;                                 // B * 96  * HW
    float* bufH    = bufLN + (size_t)Bn * DIM * HW;      // B * 576 * HW
    float* bufD    = bufH  + (size_t)Bn * 576 * HW;      // B * 576 * HW
    float* wPadIn  = bufD  + (size_t)Bn * 576 * HW;      // 512 * 96
    float* wPadOut = wPadIn + (size_t)512 * 96;          // 96 * 256

    dim3 blk(256);

    // 0) pad FFN weights (tiny)
    pad_ffn_in_w <<<(512 * 96 + 255) / 256, blk, 0, stream>>>(ffn_in_w,  wPadIn);
    pad_ffn_out_w<<<(96 * 256 + 255) / 256, blk, 0, stream>>>(ffn_out_w, wPadOut);

    // 1) LN1
    ln_biasfree<<<(Bn * HW) / 256, blk, 0, stream>>>(x, norm1_w, bufLN, DIM, HW);
    // 2) to_hidden 1x1 conv (576 x 96)
    gemm1x1_wmma<96, 576, false><<<dim3(HW / 256, 576 / 32, Bn), blk, 0, stream>>>(
        bufLN, to_hidden_w, nullptr, bufH);
    // 3) depthwise 3x3 on 576 channels
    long t3 = (long)Bn * 576 * HW;
    dwconv3<<<(int)((t3 + 255) / 256), blk, 0, stream>>>(bufH, to_hidden_dw_w, bufD,
                                                         576, 576, H, W, t3);
    // 4) windowed FFT attention -> (B, 192, H, W) into bufH (free now)
    attn_fft_kernel<<<Bn * HEADS * (H / WINP) * (W / WINP), blk, 0, stream>>>(
        bufD, temperature, bufH, H, W);
    // 5) project_out (96 x 192) + residual x -> d_out (= x1)
    gemm1x1_wmma<192, 96, true><<<dim3(HW / 256, 96 / 32, Bn), blk, 0, stream>>>(
        bufH, project_out_w, x, out);
    // 6) LN2
    ln_biasfree<<<(Bn * HW) / 256, blk, 0, stream>>>(out, norm2_w, bufLN, DIM, HW);
    // 7) ffn_in 1x1 conv (512 x 96 padded) -> bufH as (B, 512, HW)
    gemm1x1_wmma<96, 512, false><<<dim3(HW / 256, 512 / 32, Bn), blk, 0, stream>>>(
        bufLN, wPadIn, nullptr, bufH);
    // 8) depthwise 3x3 on 510 logical channels (stride 512)
    long t8 = (long)Bn * 510 * HW;
    dwconv3<<<(int)((t8 + 255) / 256), blk, 0, stream>>>(bufH, ffn_dw_w, bufD,
                                                         510, 512, H, W, t8);
    // 9) GELU gate -> bufH as (B, 256, HW), channel 255 zeroed
    long t9 = (long)Bn * 256 * HW;
    gelu_gate<<<(int)((t9 + 255) / 256), blk, 0, stream>>>(bufD, bufH, t9);
    // 10) ffn_out (96 x 256 padded) + residual x1 (in place on d_out)
    gemm1x1_wmma<256, 96, true><<<dim3(HW / 256, 96 / 32, Bn), blk, 0, stream>>>(
        bufH, wPadOut, out, out);
}
